// DGLGATNE_81879256531120
// MI455X (gfx1250) — matrix operand order repaired and checked by reading, not compiled, not run on hardware
//
#include <hip/hip_runtime.h>
#include <hip/hip_fp16.h>

// MI455X / gfx1250 fused DGL-GATNE inference kernel.
//
// kernel 1 (prep): pre-swizzle trans_weights (T,U,EMB) f32 into f16 hi/lo
//   planes laid out exactly as v_wmma_f32_16x16x32_f16 B-fragments
//   (per (ty,tile,lane): 16 contiguous halves), zero-padded for the
//   partial last tile. Lives in d_ws; rebuilt every launch (deterministic).
//
// kernel 2 (main): one workgroup (256 threads = 8 wave32) per 16 batch rows:
//   stage 1: coalesced random gather+sum of neighbor type-embeddings -> LDS
//   stage 2: attention logits (32x32 matvec + tanh) from LDS-staged tw1/tw2
//   stage 3: softmax over T=4, comb = att . nte, split-f16 (hi+residual)
//   stage 4: out = sum_ty maskA(ty) @ W[ty] via v_wmma_f32_16x16x32_f16,
//            3 WMMAs per (ty,tile): Ah*Bh + Ah*Bl + Al*Bh ~ f32 precision;
//            B fragments are two 32B coalesced loads from the prep buffer.
//   stage 5: += gathered node_embeddings, L2-normalize, store.

typedef __attribute__((ext_vector_type(16))) _Float16 v16h;
typedef __attribute__((ext_vector_type(8)))  float    v8f;
typedef __attribute__((ext_vector_type(4)))  float    v4f;

#define EMB    200
#define U_DIM  32
#define T_DIM  4
#define A_DIM  32
#define NN     10
#define BM     16
#define NTILES 13   // ceil(EMB/16)

// halves per B plane: T * NTILES * 32 lanes * 16 halves
#define WPREP_HALVES ((size_t)T_DIM * NTILES * 32 * 16)
#define WPREP_BYTES  (2 * WPREP_HALVES * sizeof(_Float16))

// ---------------------------------------------------------------------------
// Prep: swizzle trans_w into WMMA B-fragment layout, f16 hi + residual lo.
// B-layout used here (contiguous k order, matched by A): lane -> N = lane&15,
//   half e -> K = (lane>>4)*16 + e. One block per (ty, tile).
// ---------------------------------------------------------------------------
__global__ __launch_bounds__(256) void gatne_wprep_kernel(
    const float* __restrict__ trans_w,   // (T,U,EMB)
    _Float16*    __restrict__ whi,
    _Float16*    __restrict__ wlo)
{
    const int ty = blockIdx.x / NTILES;
    const int tt = blockIdx.x % NTILES;
    const float* wbase = trans_w + (size_t)ty * (U_DIM * EMB);

    for (int idx = threadIdx.x; idx < 32 * 16; idx += 256) {
        const int lane = idx >> 4;           // 0..31
        const int e    = idx & 15;           // 0..15
        const int k    = (lane >> 4) * 16 + e;      // 0..31
        const int ncol = tt * 16 + (lane & 15);
        const float w  = (ncol < EMB) ? wbase[(size_t)k * EMB + ncol] : 0.0f;
        const _Float16 wh = (_Float16)w;
        const size_t off = ((size_t)(ty * NTILES + tt) * 32 + lane) * 16 + e;
        whi[off] = wh;
        wlo[off] = (_Float16)(w - (float)wh);
    }
}

// ---------------------------------------------------------------------------
// Main fused kernel.
// ---------------------------------------------------------------------------
__global__ __launch_bounds__(256) void gatne_fused_kernel(
    const int*   __restrict__ train_inputs,   // (B,)
    const int*   __restrict__ train_types,    // (B,)
    const int*   __restrict__ node_neigh,     // (B,T,NN)
    const float* __restrict__ node_emb,       // (NUM_NODES,EMB)
    const float* __restrict__ node_type_emb,  // (NUM_NODES,T,U)
    const float* __restrict__ trans_w,        // (T,U,EMB)
    const float* __restrict__ trans_w1,       // (T,U,A)
    const float* __restrict__ trans_w2,       // (T,A,1)
    const _Float16* __restrict__ whi,         // prepped B hi (or null)
    const _Float16* __restrict__ wlo,         // prepped B lo (or null)
    float*       __restrict__ out)            // (B,EMB)
{
    __shared__ float    s_nte[BM][T_DIM][U_DIM];     // 8 KB
    __shared__ float    s_tw1[T_DIM][U_DIM][A_DIM];  // 16 KB
    __shared__ float    s_tw2[T_DIM][A_DIM];         // 512 B
    __shared__ _Float16 s_ah[BM][U_DIM];             // comb hi (f16)
    __shared__ _Float16 s_al[BM][U_DIM];             // comb residual (f16)
    __shared__ float    s_out[BM][EMB];              // 12.5 KB
    __shared__ float    s_scores[BM][T_DIM];
    __shared__ float    s_att[BM][T_DIM];
    __shared__ float    s_red[BM][16];
    __shared__ int      s_type[BM];
    __shared__ int      s_gidx[BM];
    __shared__ int      s_typemask;

    const int tid = threadIdx.x;
    const int b0  = blockIdx.x * BM;

    if (tid < BM) {
        const int gi = train_inputs[b0 + tid];
        s_gidx[tid] = gi;
        s_type[tid] = train_types[b0 + tid];
        // Warm L2/L1 for the node_embeddings rows consumed in stage 5.
        const float* p = node_emb + (size_t)gi * EMB;
        #pragma unroll
        for (int off = 0; off < EMB; off += 32)
            __builtin_prefetch(p + off, 0, 0);   // -> global_prefetch_b8
    }
    if (tid == 0) s_typemask = 0;
    // Stage attention weights in LDS (16 KB + 512 B, coalesced).
    for (int i = tid; i < T_DIM * U_DIM * A_DIM; i += 256)
        (&s_tw1[0][0][0])[i] = trans_w1[i];
    for (int i = tid; i < T_DIM * A_DIM; i += 256)
        (&s_tw2[0][0])[i] = trans_w2[i];
    __syncthreads();
    if (tid < BM) atomicOr(&s_typemask, 1 << s_type[tid]);

    // ---- Stage 1: neighbor gather + sum over NN=10 ----------------------
    // 64 (b,t) pairs, 4 lanes per pair each covering 8 of the 32 u's:
    // each neighbor row is one coalesced 128 B read.
    {
        const int pair = tid >> 2;        // 0..63
        const int q    = tid & 3;         // u-chunk (8 floats)
        const int bl   = pair >> 2;       // local row 0..15
        const int t    = pair & 3;
        float acc[8];
        #pragma unroll
        for (int j = 0; j < 8; ++j) acc[j] = 0.0f;
        const int nbase = (b0 + bl) * (T_DIM * NN) + t * NN;
        #pragma unroll
        for (int n = 0; n < NN; ++n) {
            const long node = node_neigh[nbase + n];
            const float* src = node_type_emb + ((size_t)node * T_DIM + t) * U_DIM + q * 8;
            const v4f lo = *(const v4f*)(src);
            const v4f hi = *(const v4f*)(src + 4);
            #pragma unroll
            for (int j = 0; j < 4; ++j) { acc[j] += lo[j]; acc[4 + j] += hi[j]; }
        }
        #pragma unroll
        for (int j = 0; j < 8; ++j) s_nte[bl][t][q * 8 + j] = acc[j];
    }
    __syncthreads();

    // ---- Stage 2: attention logits, one thread per (b,t) ----------------
    if (tid < BM * T_DIM) {
        const int bl = tid >> 2, t = tid & 3;
        const int ty = s_type[bl];
        float score = 0.0f;
        for (int a = 0; a < A_DIM; ++a) {
            float hsum = 0.0f;
            #pragma unroll
            for (int u = 0; u < U_DIM; ++u)
                hsum += s_nte[bl][t][u] * s_tw1[ty][u][a];
            score += tanhf(hsum) * s_tw2[ty][a];
        }
        s_scores[bl][t] = score;
    }
    __syncthreads();

    // ---- Stage 3a: softmax over T=4 -------------------------------------
    if (tid < BM) {
        float m = s_scores[tid][0];
        #pragma unroll
        for (int t = 1; t < T_DIM; ++t) m = fmaxf(m, s_scores[tid][t]);
        float e[T_DIM], den = 0.0f;
        #pragma unroll
        for (int t = 0; t < T_DIM; ++t) { e[t] = __expf(s_scores[tid][t] - m); den += e[t]; }
        const float inv = 1.0f / den;
        #pragma unroll
        for (int t = 0; t < T_DIM; ++t) s_att[tid][t] = e[t] * inv;
    }
    __syncthreads();

    // ---- Stage 3b: comb = att . nte, split into f16 hi + residual -------
    if (tid < BM * 8) {
        const int bl = tid >> 3;
        const int u0 = (tid & 7) * 4;
        #pragma unroll
        for (int j = 0; j < 4; ++j) {
            const int u = u0 + j;
            float c = 0.0f;
            #pragma unroll
            for (int t = 0; t < T_DIM; ++t) c += s_att[bl][t] * s_nte[bl][t][u];
            const _Float16 ch = (_Float16)c;
            s_ah[bl][u] = ch;
            s_al[bl][u] = (_Float16)(c - (float)ch);
        }
    }
    __syncthreads();

    // ---- Stage 4: out_tile = sum_ty maskA(ty) @ W[ty] via WMMA ----------
    // A-layout (16-bit A 16x32): lane = {khalf(1),m(4)};
    //   half e -> K = khalf*16+e (matches B fragment K order).
    // B fragments: prepped in d_ws (two 32B loads) or inline fallback.
    // D-layout (f32 C/D): lane = N, VGPR r -> M = r + 8*(lane>>4).
    {
        const int wave  = tid >> 5;     // 0..7
        const int lane  = tid & 31;
        const int mrow  = lane & 15;
        const int lhalf = lane >> 4;
        const int rowty = s_type[mrow];
        const int tmask = s_typemask;

        for (int tt = wave; tt < NTILES; tt += 8) {
            const int n0   = tt * 16;
            const int ncol = n0 + (lane & 15);
            v8f c;
            #pragma unroll
            for (int r = 0; r < 8; ++r) c[r] = 0.0f;

            for (int ty = 0; ty < T_DIM; ++ty) {
                if (!((tmask >> ty) & 1)) continue;   // workgroup-uniform skip

                v16h ah, al;
                #pragma unroll
                for (int e = 0; e < 16; ++e) { ah[e] = (_Float16)0; al[e] = (_Float16)0; }
                if (rowty == ty) {                    // row-type mask on A
                    #pragma unroll
                    for (int e = 0; e < 8; ++e) {
                        ah[e]     = s_ah[mrow][lhalf * 16 + e];
                        ah[8 + e] = s_ah[mrow][lhalf * 16 + 8 + e];
                        al[e]     = s_al[mrow][lhalf * 16 + e];
                        al[8 + e] = s_al[mrow][lhalf * 16 + 8 + e];
                    }
                }

                v16h bh, blo;
                if (whi != nullptr) {
                    // Pre-swizzled B fragment: 16 contiguous halves per lane
                    // -> two coalesced global_load_b128 per plane.
                    const size_t off = ((size_t)(ty * NTILES + tt) * 32 + lane) * 16;
                    bh  = *(const v16h*)(whi + off);
                    blo = *(const v16h*)(wlo + off);
                } else {
                    // Fallback: inline strided loads + split.
                    const float* wbase = trans_w + (size_t)ty * (U_DIM * EMB);
                    #pragma unroll
                    for (int e = 0; e < 16; ++e) {
                        const int k = lhalf * 16 + e;
                        const float w = (ncol < EMB) ? wbase[(size_t)k * EMB + ncol] : 0.0f;
                        const _Float16 wh = (_Float16)w;
                        bh[e]  = wh;
                        blo[e] = (_Float16)(w - (float)wh);
                    }
                }

                // D = A*B + C, f32 accumulate (split-f16 ~ f32 precision)
                c = __builtin_amdgcn_wmma_f32_16x16x32_f16(false, ah, false, bh,  (short)0, c, false, false);
                c = __builtin_amdgcn_wmma_f32_16x16x32_f16(false, ah, false, blo, (short)0, c, false, false);
                c = __builtin_amdgcn_wmma_f32_16x16x32_f16(false, al, false, bh,  (short)0, c, false, false);
            }

            if (ncol < EMB) {
                #pragma unroll
                for (int r = 0; r < 8; ++r)
                    s_out[8 * lhalf + r][ncol] = c[r];
            }
        }
    }
    __syncthreads();

    // ---- Stage 5: += node_embeddings gather, L2 normalize, store --------
    {
        const int row = tid >> 4;        // 0..15
        const int c0  = tid & 15;
        const float* ne = node_emb + (size_t)s_gidx[row] * EMB;
        float ss = 0.0f;
        for (int cidx = c0; cidx < EMB; cidx += 16) {
            const float v = s_out[row][cidx] + ne[cidx];
            s_out[row][cidx] = v;
            ss += v * v;
        }
        s_red[row][c0] = ss;
    }
    __syncthreads();
    if (tid < BM) {
        float s = 0.0f;
        #pragma unroll
        for (int j = 0; j < 16; ++j) s += s_red[tid][j];
        s_red[tid][0] = 1.0f / fmaxf(sqrtf(s), 1e-12f);
    }
    __syncthreads();
    {
        const int row = tid >> 4;
        const int c0  = tid & 15;
        const float inv = s_red[row][0];
        float* dst = out + (size_t)(b0 + row) * EMB;
        for (int cidx = c0; cidx < EMB; cidx += 16)
            dst[cidx] = s_out[row][cidx] * inv;
    }
}

extern "C" void kernel_launch(void* const* d_in, const int* in_sizes, int n_in,
                              void* d_out, int out_size, void* d_ws, size_t ws_size,
                              hipStream_t stream) {
    const int*   train_inputs         = (const int*)d_in[0];
    const int*   train_types          = (const int*)d_in[1];
    const int*   node_neigh           = (const int*)d_in[2];
    const float* node_embeddings      = (const float*)d_in[3];
    const float* node_type_embeddings = (const float*)d_in[4];
    const float* trans_weights        = (const float*)d_in[5];
    const float* trans_weights_s1     = (const float*)d_in[6];
    const float* trans_weights_s2     = (const float*)d_in[7];
    float* outp = (float*)d_out;

    _Float16* whi = nullptr;
    _Float16* wlo = nullptr;
    if (ws_size >= WPREP_BYTES) {
        whi = (_Float16*)d_ws;
        wlo = whi + WPREP_HALVES;
        gatne_wprep_kernel<<<dim3(T_DIM * NTILES), dim3(256), 0, stream>>>(
            trans_weights, whi, wlo);
    }

    const int Bsz = in_sizes[0];           // 16384
    dim3 grid(Bsz / BM), block(256);
    gatne_fused_kernel<<<grid, block, 0, stream>>>(
        train_inputs, train_types, node_neigh,
        node_embeddings, node_type_embeddings,
        trans_weights, trans_weights_s1, trans_weights_s2,
        whi, wlo, outp);
    (void)n_in; (void)out_size;
}
